// OutputLayer_69269232549995
// MI455X (gfx1250) — matrix operand order repaired and checked by reading, not compiled, not run on hardware
//
#include <hip/hip_runtime.h>
#include <hip/hip_bf16.h>

// Problem constants (match reference)
#define BB   64
#define NN   2048
#define DD   512
#define HH   512
#define MROWS (BB * NN)        // 131072
#define CA   3
#define CB   5
#define BN_EPS 1e-5f

#define TILE_M 128
#define TILE_N 128
#define TILE_K 32

typedef __attribute__((ext_vector_type(16))) __bf16 v16bf;
typedef __attribute__((ext_vector_type(8)))  float  v8f;

// ---------------------------------------------------------------------------
// Kernel 1: convert W1 (f32, [D][H] row-major) -> bf16
// ---------------------------------------------------------------------------
__global__ __launch_bounds__(256)
void cvt_w1_kernel(const float* __restrict__ w1, __bf16* __restrict__ w1bf) {
    int i = blockIdx.x * 256 + threadIdx.x;      // grid sized exactly D*H/256
    w1bf[i] = (__bf16)w1[i];
}

// ---------------------------------------------------------------------------
// Kernel 2: zero the stats buffer (sum[H], sumsq[H])
// ---------------------------------------------------------------------------
__global__ __launch_bounds__(256)
void zero_stats_kernel(float* __restrict__ stats) {
    int i = blockIdx.x * 256 + threadIdx.x;      // grid covers 2*H floats
    if (i < 2 * HH) stats[i] = 0.0f;
}

// ---------------------------------------------------------------------------
// Kernel 3: h = x @ W1 + b1 (BF16 WMMA, f32 accum), h stored as bf16,
//           per-feature sum/sumsq accumulated into stats via atomics.
//
//   grid = (HH/TILE_N, MROWS/TILE_M), block = 256 (8 wave32).
//   N-tiles are the FAST grid dimension: the 4 blocks sharing the same
//   128 rows of x are consecutive in dispatch order, so their x strip
//   (128x512x4B = 256 KB) is served from the 192 MB L2 after the first
//   touch. x then streams from HBM once (~268 MB) instead of 4x.
//
//   wave layout: 4 (m) x 2 (n); each wave owns 32x64 => 2x4 WMMA frags.
// ---------------------------------------------------------------------------
__global__ __launch_bounds__(256)
void gemm_bn_stats_kernel(const float*  __restrict__ x,
                          const __bf16* __restrict__ w1bf,
                          const float*  __restrict__ b1,
                          __bf16*       __restrict__ hbf,
                          float*        __restrict__ stats) {
    __shared__ __bf16 As[TILE_M * TILE_K];   // [m][k], row-major, 8 KB
    __shared__ __bf16 Bs[TILE_N * TILE_K];   // [n][k], K-contiguous, 8 KB
    __shared__ float  s_sum[TILE_N];
    __shared__ float  s_ssq[TILE_N];

    const int t    = threadIdx.x;
    const int lane = t & 31;
    const int wave = t >> 5;          // 0..7
    const int wm   = wave >> 1;       // 0..3  (row strip of 32)
    const int wn   = wave & 1;        // 0..1  (col strip of 64)
    const int half = lane >> 4;       // 0/1
    const int l15  = lane & 15;

    const int nbase = blockIdx.x * TILE_N;   // fast dim -> L2 reuse of x
    const int mbase = blockIdx.y * TILE_M;

    if (t < TILE_N) { s_sum[t] = 0.0f; s_ssq[t] = 0.0f; }

    v8f acc[2][4] = {};

    // A tile cooperative-load mapping: 2 threads per row, 16 K-elems each
    const int arow  = t >> 1;            // 0..127
    const int akseg = (t & 1) * 16;      // 0 or 16
    // B tile mapping: 8 threads per K-row, 16 N-elems each
    const int bk    = t >> 3;            // 0..31
    const int bnseg = (t & 7) * 16;      // 0,16,...,112

    for (int k0 = 0; k0 < DD; k0 += TILE_K) {
        __syncthreads();

        // ---- stage A tile: f32 global -> bf16 LDS ----
        {
            const float4* src = reinterpret_cast<const float4*>(
                x + (size_t)(mbase + arow) * DD + k0 + akseg);
            float4 f[4];
            f[0] = src[0]; f[1] = src[1]; f[2] = src[2]; f[3] = src[3];
            const float* fe = reinterpret_cast<const float*>(f);
            __bf16* dst = &As[arow * TILE_K + akseg];
            #pragma unroll
            for (int i = 0; i < 16; ++i) dst[i] = (__bf16)fe[i];
        }
        // prefetch next x tile (global_prefetch_b8)
        if (k0 + TILE_K < DD)
            __builtin_prefetch(x + (size_t)(mbase + arow) * DD + (k0 + TILE_K) + akseg, 0, 0);

        // ---- stage B tile: bf16 global (L2-resident) -> LDS, [n][k] ----
        {
            const uint4* src = reinterpret_cast<const uint4*>(
                w1bf + (size_t)(k0 + bk) * HH + nbase + bnseg);
            uint4 u[2];
            u[0] = src[0]; u[1] = src[1];
            const __bf16* e = reinterpret_cast<const __bf16*>(u);
            #pragma unroll
            for (int i = 0; i < 16; ++i)
                Bs[(bnseg + i) * TILE_K + bk] = e[i];
        }

        __syncthreads();

        // ---- load fragments, contiguous K order (k = half*16 + s) for A and B ----
        union AV { v16bf v; unsigned u[8]; };
        AV a[2], b[4];

        #pragma unroll
        for (int mf = 0; mf < 2; ++mf) {
            const int m = wm * 32 + mf * 16 + l15;
            #pragma unroll
            for (int v = 0; v < 8; ++v) {
                // A 16x32: vgpr0-7 -> K = half*16 + {0..15}
                const int kb = half * 16 + 2 * v;
                a[mf].u[v] = *reinterpret_cast<const unsigned*>(&As[m * TILE_K + kb]);
            }
        }
        #pragma unroll
        for (int nf = 0; nf < 4; ++nf) {
            // B 32x16: lanes0-15 K=0..15, lanes16-31 K=16..31, col = lane&15
            const int n = wn * 64 + nf * 16 + l15;
            const unsigned* bp =
                reinterpret_cast<const unsigned*>(&Bs[n * TILE_K + half * 16]);
            #pragma unroll
            for (int v = 0; v < 8; ++v) b[nf].u[v] = bp[v];
        }

        // ---- 8 WMMAs per wave per K-step ----
        #pragma unroll
        for (int mf = 0; mf < 2; ++mf)
            #pragma unroll
            for (int nf = 0; nf < 4; ++nf)
                acc[mf][nf] = __builtin_amdgcn_wmma_f32_16x16x32_bf16(
                    false, a[mf].v, false, b[nf].v,
                    (short)0, acc[mf][nf], false, false);
    }

    // ---- epilogue: +bias, store h (bf16), accumulate BN stats ----
    #pragma unroll
    for (int mf = 0; mf < 2; ++mf) {
        #pragma unroll
        for (int nf = 0; nf < 4; ++nf) {
            const int nl = wn * 64 + nf * 16 + l15;   // col within tile
            const int ng = nbase + nl;                // global col
            const float bv = b1[ng];
            float lsum = 0.0f, lssq = 0.0f;
            #pragma unroll
            for (int r = 0; r < 8; ++r) {
                // C/D layout: vgpr r -> M = r + half*8; N = lane&15
                const int ml = wm * 32 + mf * 16 + half * 8 + r;
                const float val = acc[mf][nf][r] + bv;
                hbf[(size_t)(mbase + ml) * HH + ng] = (__bf16)val;
                lsum += val;
                lssq += val * val;
            }
            atomicAdd(&s_sum[nl], lsum);   // ds_add_f32
            atomicAdd(&s_ssq[nl], lssq);
        }
    }
    __syncthreads();
    if (t < TILE_N) {
        atomicAdd(&stats[nbase + t],      s_sum[t]);   // global_atomic_add_f32
        atomicAdd(&stats[HH + nbase + t], s_ssq[t]);
    }
}

// ---------------------------------------------------------------------------
// Kernel 4: fold BN stats + gamma/beta into per-feature scale/shift
// ---------------------------------------------------------------------------
__global__ __launch_bounds__(256)
void bn_scale_kernel(const float* __restrict__ stats,
                     const float* __restrict__ gamma,
                     const float* __restrict__ beta,
                     float* __restrict__ scale,
                     float* __restrict__ shift) {
    int j = blockIdx.x * 256 + threadIdx.x;
    if (j < HH) {
        const float inv = 1.0f / (float)MROWS;
        float mu  = stats[j] * inv;
        float var = stats[HH + j] * inv - mu * mu;
        float rs  = rsqrtf(var + BN_EPS);
        float g   = gamma[j] * rs;
        scale[j]  = g;
        shift[j]  = beta[j] - mu * g;
    }
}

// ---------------------------------------------------------------------------
// Kernel 5: normalize + ReLU + head logits + argmax. One wave per row.
//   Row r: node index n = r % NN; n < NN/2 -> head A (3 cls) else B (5 cls).
//   scale/shift (4 KB) are staged in LDS once per block.
// ---------------------------------------------------------------------------
__global__ __launch_bounds__(256)
void head_kernel(const __bf16* __restrict__ hbf,
                 const float*  __restrict__ scale,
                 const float*  __restrict__ shift,
                 const float*  __restrict__ Wa, const float* __restrict__ ba,
                 const float*  __restrict__ Wb, const float* __restrict__ bb,
                 float*        __restrict__ out) {
    __shared__ float s_scale[HH];
    __shared__ float s_shift[HH];
    {
        const int t = threadIdx.x;
        #pragma unroll
        for (int r = 0; r < HH / 256; ++r) {
            s_scale[t + r * 256] = scale[t + r * 256];
            s_shift[t + r * 256] = shift[t + r * 256];
        }
    }
    __syncthreads();

    const int lane = threadIdx.x & 31;
    const int row  = blockIdx.x * 8 + (threadIdx.x >> 5);
    const int n    = row & (NN - 1);
    const bool isA = n < (NN / 2);
    const int C            = isA ? CA : CB;
    const float* __restrict__ W    = isA ? Wa : Wb;
    const float* __restrict__ bias = isA ? ba : bb;

    const int j0 = lane * 16;
    uint4 u[2];
    const uint4* src = reinterpret_cast<const uint4*>(hbf + (size_t)row * HH + j0);
    u[0] = src[0]; u[1] = src[1];
    const __bf16* e = reinterpret_cast<const __bf16*>(u);

    float acc[CB] = {0.f, 0.f, 0.f, 0.f, 0.f};
    #pragma unroll
    for (int i = 0; i < 16; ++i) {
        const int j = j0 + i;
        float v = (float)e[i] * s_scale[j] + s_shift[j];
        v = v > 0.0f ? v : 0.0f;                      // ReLU
        #pragma unroll
        for (int c = 0; c < CB; ++c)
            if (c < C) acc[c] += v * W[j * C + c];
    }
    // wave32 reduction
    #pragma unroll
    for (int c = 0; c < CB; ++c)
        #pragma unroll
        for (int off = 16; off > 0; off >>= 1)
            acc[c] += __shfl_xor(acc[c], off, 32);

    if (lane == 0) {
        // argmax of logits == argmax of softmax; first-max tie-break
        float best = acc[0] + bias[0];
        int bi = 0;
        #pragma unroll
        for (int c = 1; c < CB; ++c) {
            if (c < C) {
                float l = acc[c] + bias[c];
                if (l > best) { best = l; bi = c; }
            }
        }
        out[row] = (float)bi;
    }
}

// ---------------------------------------------------------------------------
// Launch
// ---------------------------------------------------------------------------
extern "C" void kernel_launch(void* const* d_in, const int* in_sizes, int n_in,
                              void* d_out, int out_size, void* d_ws, size_t ws_size,
                              hipStream_t stream) {
    (void)in_sizes; (void)n_in; (void)out_size; (void)ws_size;

    const float* x     = (const float*)d_in[0];
    const float* W1    = (const float*)d_in[1];
    const float* b1    = (const float*)d_in[2];
    const float* gamma = (const float*)d_in[3];
    const float* beta  = (const float*)d_in[4];
    const float* Wa    = (const float*)d_in[5];
    const float* ba    = (const float*)d_in[6];
    const float* Wb    = (const float*)d_in[7];
    const float* bb    = (const float*)d_in[8];
    // d_in[9], d_in[10] (idx_a / idx_b) are contiguous ranges [0,N/2) and
    // [N/2,N); handled analytically in head_kernel.

    // Workspace layout
    char* ws = (char*)d_ws;
    __bf16* hbf   = (__bf16*)ws;                                   // 134,217,728 B
    __bf16* w1bf  = (__bf16*)(ws + (size_t)MROWS * HH * 2);        //     524,288 B
    float*  stats = (float*)(ws + (size_t)MROWS * HH * 2 + (size_t)DD * HH * 2);
    float*  scale = stats + 2 * HH;
    float*  shift = scale + HH;

    cvt_w1_kernel<<<(DD * HH) / 256, 256, 0, stream>>>(W1, w1bf);
    zero_stats_kernel<<<(2 * HH + 255) / 256, 256, 0, stream>>>(stats);
    // N-tiles fastest => sibling tiles of one M-strip dispatch back-to-back
    gemm_bn_stats_kernel<<<dim3(HH / TILE_N, MROWS / TILE_M), 256, 0, stream>>>(
        x, w1bf, b1, hbf, stats);
    bn_scale_kernel<<<(HH + 255) / 256, 256, 0, stream>>>(stats, gamma, beta, scale, shift);
    head_kernel<<<MROWS / 8, 256, 0, stream>>>(hbf, scale, shift, Wa, ba, Wb, bb,
                                               (float*)d_out);
}
